// PainnUpdate_41291815584027
// MI455X (gfx1250) — matrix operand order repaired and checked by reading, not compiled, not run on hardware
//
#include <hip/hip_runtime.h>
#include <hip/hip_bf16.h>

#define F 128
#define ROW (4*F)          // 512 floats per node row

typedef __attribute__((ext_vector_type(16))) __bf16 v16bf;
typedef __attribute__((ext_vector_type(8)))  __bf16 v8bf;
typedef __attribute__((ext_vector_type(8)))  float  v8f;
typedef __attribute__((ext_vector_type(4)))  float  f32x4;

// ---- workspace layout (bf16 elements): B-fragments [kt][nt][lane][16] ------
#define WS_U   0          // U_w  : 4*8  frags (16384 elems)
#define WS_V   16384      // V_w  : 4*8  frags
#define WS_M1  32768      // m1_w : 8*8  frags (32768 elems)
#define WS_M2  65536      // m2_w : 4*24 frags (49152 elems)
#define WS_TOTAL 114688

// ---------------------------------------------------------------------------
// Weight prep: f32 row-major (K x Ncols) -> bf16 WMMA B-fragments.
// B layout (16-bit, 32x16): n = lane&15, g = lane>>4; elems j=0..7 -> K = g*8+j,
// elems j=8..15 -> K = 16+g*8+(j-8) (same k order as A).
// ---------------------------------------------------------------------------
__device__ __forceinline__ void prep_one(const float* __restrict__ W,
                                         __bf16* __restrict__ dst,
                                         int idx, int n_tiles, int ncols) {
  int frag = idx >> 9;
  int r    = idx & 511;
  int lane = r >> 4;
  int j    = r & 15;
  int kt = frag / n_tiles;
  int nt = frag - kt * n_tiles;
  int k = kt * 32 + ((j & 8) << 1) + ((lane >> 4) << 3) + (j & 7);
  int n = nt * 16 + (lane & 15);
  dst[idx] = (__bf16)W[(size_t)k * ncols + n];
}

__global__ void __launch_bounds__(256)
prep_weights(const float* __restrict__ Uw, const float* __restrict__ Vw,
             const float* __restrict__ m1w, const float* __restrict__ m2w,
             __bf16* __restrict__ ws) {
  int idx = blockIdx.x * 256 + threadIdx.x;
  if (idx < WS_V)           prep_one(Uw,  ws + WS_U,  idx - WS_U,   8, F);
  else if (idx < WS_M1)     prep_one(Vw,  ws + WS_V,  idx - WS_V,   8, F);
  else if (idx < WS_M2)     prep_one(m1w, ws + WS_M1, idx - WS_M1,  8, F);
  else if (idx < WS_TOTAL)  prep_one(m2w, ws + WS_M2, idx - WS_M2, 24, 3 * F);
}

// ---------------------------------------------------------------------------
// A-fragment builders.
// A layout (16-bit, 16x32): lane m = l&15, group g = l>>4,
//   elems j=0..7 -> K = g*8+j ; elems j=8..15 -> K = 16+g*8+(j-8)
// ---------------------------------------------------------------------------
__device__ __forceinline__ v16bf afrag_bf16(const __bf16* __restrict__ p) {
  // p = row base + kbase + g*8 ; two contiguous 16-byte chunks
  v8bf lo = *(const v8bf*)(p);
  v8bf hi = *(const v8bf*)(p + 16);
  v16bf a;
#pragma unroll
  for (int j = 0; j < 8; ++j) { a[j] = lo[j]; a[j + 8] = hi[j]; }
  return a;
}

__device__ __forceinline__ v16bf afrag_f32g(const float* __restrict__ p) {
  v16bf a;
#pragma unroll
  for (int j = 0; j < 8; ++j) {
    a[j]     = (__bf16)p[j];
    a[j + 8] = (__bf16)p[16 + j];
  }
  return a;
}

__device__ __forceinline__ v16bf wfrag(const __bf16* __restrict__ w,
                                       int fragIdx, int lane) {
  return *(const v16bf*)(w + ((size_t)fragIdx << 9) + (lane << 4));
}

__device__ __forceinline__ v8f wmma_bf16(v16bf a, v16bf b, v8f c) {
  return __builtin_amdgcn_wmma_f32_16x16x32_bf16(false, a, false, b,
                                                 (short)0, c, false, false);
}

// ---------------------------------------------------------------------------
// Main kernel: one wave (32 threads) per block == one 32-node tile
// (two 16-row WMMA A-tiles sharing every B-fragment load).
// Dynamic LDS (64 KB):
//   s_uv  [3][32][128] bf16 (24 KB)  Uv staged for delta_v
//   s_vo  [32][384]    bf16 (24 KB)  vector input, later mlp_out
//   s_act [32][128]    bf16 ( 8 KB)  |Vv| then h
//   s_inn [32][128]    bf16 ( 8 KB)  inner product sum_c Uv*Vv
// ---------------------------------------------------------------------------
__global__ void __launch_bounds__(32)
painn_update(const float* __restrict__ nf,
             const float* __restrict__ Ub,  const float* __restrict__ Vb,
             const float* __restrict__ m1b, const float* __restrict__ m2b,
             const __bf16* __restrict__ ws,
             float* __restrict__ out) {
  extern __shared__ char smem[];
  __bf16* s_uv  = (__bf16*)(smem);
  __bf16* s_vo  = (__bf16*)(smem + 24576);
  __bf16* s_act = (__bf16*)(smem + 49152);
  __bf16* s_inn = (__bf16*)(smem + 57344);

  const int lane = threadIdx.x;
  const int g  = lane >> 4;
  const int nl = lane & 15;
  const int n0 = blockIdx.x * 32;

  const float* nfrow  = nf  + (size_t)n0 * ROW;
  float*       outrow = out + (size_t)n0 * ROW;
  const __bf16 *wsU = ws + WS_U, *wsV = ws + WS_V;
  const __bf16 *wsM1 = ws + WS_M1, *wsM2 = ws + WS_M2;

  // ---- stage vector features (cols 128..511) as bf16 into s_vo ------------
#pragma unroll 1
  for (int m = 0; m < 32; ++m) {
#pragma unroll
    for (int q = 0; q < 3; ++q) {
      int col = (q * 32 + lane) * 4;
      f32x4 x = *(const f32x4*)(nfrow + (size_t)m * ROW + F + col);
      __bf16* d = s_vo + m * 384 + col;
      d[0] = (__bf16)x[0]; d[1] = (__bf16)x[1];
      d[2] = (__bf16)x[2]; d[3] = (__bf16)x[3];
    }
  }

  // ---- Pass 1: Uv, Vv, |Vv|^2, inner=sum_c Uv*Vv (nt-outer, B-frag reuse) --
#pragma unroll 1
  for (int nt = 0; nt < 8; ++nt) {
    v16bf bU[4], bV[4];
#pragma unroll
    for (int kt = 0; kt < 4; ++kt) {
      bU[kt] = wfrag(wsU, kt * 8 + nt, lane);
      bV[kt] = wfrag(wsV, kt * 8 + nt, lane);
    }
    const float ub = Ub[nt * 16 + nl];
    const float vb = Vb[nt * 16 + nl];
    const int col = nt * 16 + nl;

    float nrm2_0[8], nrm2_1[8], inn_0[8], inn_1[8];
#pragma unroll
    for (int r = 0; r < 8; ++r) {
      nrm2_0[r] = 0.f; nrm2_1[r] = 0.f; inn_0[r] = 0.f; inn_1[r] = 0.f;
    }

#pragma unroll 1
    for (int c = 0; c < 3; ++c) {
      v16bf a0[4], a1[4];
#pragma unroll
      for (int kt = 0; kt < 4; ++kt) {
        a0[kt] = afrag_bf16(s_vo + (nl)      * 384 + c * F + kt * 32 + g * 8);
        a1[kt] = afrag_bf16(s_vo + (16 + nl) * 384 + c * F + kt * 32 + g * 8);
      }
      v8f u0, u1, v0, v1;
#pragma unroll
      for (int r = 0; r < 8; ++r) { u0[r] = ub; u1[r] = ub; v0[r] = vb; v1[r] = vb; }
#pragma unroll
      for (int kt = 0; kt < 4; ++kt) {
        u0 = wmma_bf16(a0[kt], bU[kt], u0);
        u1 = wmma_bf16(a1[kt], bU[kt], u1);
        v0 = wmma_bf16(a0[kt], bV[kt], v0);
        v1 = wmma_bf16(a1[kt], bV[kt], v1);
      }
#pragma unroll
      for (int r = 0; r < 8; ++r) {
        int m0 = r + g * 8, m1 = 16 + r + g * 8;     // C layout rows
        nrm2_0[r] += v0[r] * v0[r];  nrm2_1[r] += v1[r] * v1[r];
        inn_0[r]  += u0[r] * v0[r];  inn_1[r]  += u1[r] * v1[r];
        s_uv[(c * 32 + m0) * F + col] = (__bf16)u0[r];
        s_uv[(c * 32 + m1) * F + col] = (__bf16)u1[r];
      }
    }
#pragma unroll
    for (int r = 0; r < 8; ++r) {
      int m0 = r + g * 8, m1 = 16 + r + g * 8;
      s_act[m0 * F + col] = (__bf16)sqrtf(nrm2_0[r]);
      s_act[m1 * F + col] = (__bf16)sqrtf(nrm2_1[r]);
      s_inn[m0 * F + col] = (__bf16)inn_0[r];
      s_inn[m1 * F + col] = (__bf16)inn_1[r];
    }
  }

  // ---- MLP layer 1: mlp_in = [ |Vv| | scalar ], K=256 ----------------------
  v16bf am0[8], am1[8];
#pragma unroll
  for (int kt = 0; kt < 4; ++kt) {
    am0[kt] = afrag_bf16(s_act + (nl)      * F + kt * 32 + g * 8);
    am1[kt] = afrag_bf16(s_act + (16 + nl) * F + kt * 32 + g * 8);
  }
#pragma unroll
  for (int kt = 0; kt < 4; ++kt) {
    am0[4 + kt] = afrag_f32g(nfrow + (size_t)(nl)      * ROW + kt * 32 + g * 8);
    am1[4 + kt] = afrag_f32g(nfrow + (size_t)(16 + nl) * ROW + kt * 32 + g * 8);
  }
#pragma unroll 1
  for (int nt = 0; nt < 8; ++nt) {
    const float b1 = m1b[nt * 16 + nl];
    const int col = nt * 16 + nl;
    v8f h0, h1;
#pragma unroll
    for (int r = 0; r < 8; ++r) { h0[r] = b1; h1[r] = b1; }
#pragma unroll
    for (int kt = 0; kt < 8; ++kt) {
      v16bf b = wfrag(wsM1, kt * 8 + nt, lane);
      h0 = wmma_bf16(am0[kt], b, h0);
      h1 = wmma_bf16(am1[kt], b, h1);
    }
#pragma unroll
    for (int r = 0; r < 8; ++r) {
      int m0 = r + g * 8, m1 = 16 + r + g * 8;
      float x0 = h0[r], x1 = h1[r];
      s_act[m0 * F + col] = (__bf16)(x0 / (1.0f + __expf(-x0)));  // silu
      s_act[m1 * F + col] = (__bf16)(x1 / (1.0f + __expf(-x1)));
    }
  }

  // ---- MLP layer 2: h(128) -> mlp_out(384), staged bf16 into s_vo ----------
  v16bf af0[4], af1[4];
#pragma unroll
  for (int kt = 0; kt < 4; ++kt) {
    af0[kt] = afrag_bf16(s_act + (nl)      * F + kt * 32 + g * 8);
    af1[kt] = afrag_bf16(s_act + (16 + nl) * F + kt * 32 + g * 8);
  }
#pragma unroll 1
  for (int nt = 0; nt < 24; ++nt) {
    const float b2 = m2b[nt * 16 + nl];
    const int col = nt * 16 + nl;                    // 0..383
    v8f o0, o1;
#pragma unroll
    for (int r = 0; r < 8; ++r) { o0[r] = b2; o1[r] = b2; }
#pragma unroll
    for (int kt = 0; kt < 4; ++kt) {
      v16bf b = wfrag(wsM2, kt * 24 + nt, lane);
      o0 = wmma_bf16(af0[kt], b, o0);
      o1 = wmma_bf16(af1[kt], b, o1);
    }
#pragma unroll
    for (int r = 0; r < 8; ++r) {
      int m0 = r + g * 8, m1 = 16 + r + g * 8;
      s_vo[m0 * 384 + col] = (__bf16)o0[r];          // overwrites staged input
      s_vo[m1 * 384 + col] = (__bf16)o1[r];
    }
  }

  // ---- Pass 2a: delta_v = a_vv * Uv ; residual add (flat element-wise) -----
#pragma unroll 1
  for (int c = 0; c < 3; ++c) {
#pragma unroll 4
    for (int i = 0; i < 128; ++i) {
      int e = i * 32 + lane;                         // 0..4095
      int m = e >> 7, col = e & 127;
      float u = (float)s_uv[(c * 32 + m) * F + col];
      float a = (float)s_vo[m * 384 + col];          // a_vv
      size_t gi = (size_t)m * ROW + F + c * F + col;
      outrow[gi] = nfrow[gi] + a * u;
    }
  }

  // ---- Pass 2b: delta_s = a_sv * inner + a_ss ; residual add ---------------
#pragma unroll 4
  for (int i = 0; i < 128; ++i) {
    int e = i * 32 + lane;
    int m = e >> 7, col = e & 127;
    float inner = (float)s_inn[m * F + col];
    float sv = (float)s_vo[m * 384 + 128 + col];
    float ss = (float)s_vo[m * 384 + 256 + col];
    size_t gi = (size_t)m * ROW + col;
    outrow[gi] = nfrow[gi] + sv * inner + ss;
  }
}

// ---------------------------------------------------------------------------
extern "C" void kernel_launch(void* const* d_in, const int* in_sizes, int n_in,
                              void* d_out, int out_size, void* d_ws, size_t ws_size,
                              hipStream_t stream) {
  const float* nf  = (const float*)d_in[0];
  const float* Uw  = (const float*)d_in[1];
  const float* Ubv = (const float*)d_in[2];
  const float* Vw  = (const float*)d_in[3];
  const float* Vbv = (const float*)d_in[4];
  const float* m1w = (const float*)d_in[5];
  const float* m1v = (const float*)d_in[6];
  const float* m2w = (const float*)d_in[7];
  const float* m2v = (const float*)d_in[8];
  __bf16* ws  = (__bf16*)d_ws;
  float*  out = (float*)d_out;

  // allow 64 KB dynamic LDS (idempotent; harmless under graph capture)
  (void)hipFuncSetAttribute((const void*)painn_update,
                            hipFuncAttributeMaxDynamicSharedMemorySize, 65536);

  prep_weights<<<(WS_TOTAL + 255) / 256, 256, 0, stream>>>(Uw, Vw, m1w, m2w, ws);

  int tiles = in_sizes[0] / (ROW * 32);              // 200000/32 = 6250
  painn_update<<<tiles, 32, 65536, stream>>>(nf, Ubv, Vbv, m1v, m2v, ws, out);
}
